// MambaEncoder_28716151341323
// MI455X (gfx1250) — matrix operand order repaired and checked by reading, not compiled, not run on hardware
//
#include <hip/hip_runtime.h>
#include <hip/hip_bf16.h>

typedef __bf16 bf16_t;
typedef __attribute__((ext_vector_type(16))) __bf16 v16bf;
typedef __attribute__((ext_vector_type(8)))  float  v8f;
typedef int v4i __attribute__((vector_size(16)));

#define BATCH   4
#define SEQ     2048
#define MROWS   (BATCH * SEQ)      // 8192
#define DMODEL  256
#define DINNER  512
#define DSTATE  16
#define DTRANK  16
#define DBCDIM  (DTRANK + 2 * DSTATE)  // 48

#if __has_builtin(__builtin_amdgcn_global_load_async_to_lds_b128)
#define HAVE_ASYNC_LDS 1
#endif

// ---------------------------------------------------------------------------
// f32 -> bf16 weight conversion
// ---------------------------------------------------------------------------
__global__ void cvt_bf16_kernel(const float* __restrict__ src,
                                bf16_t* __restrict__ dst, int n) {
    int i = blockIdx.x * blockDim.x + threadIdx.x;
    if (i < n) dst[i] = (bf16_t)src[i];
}

// ---------------------------------------------------------------------------
// embedding: h[m,d] = x[m] * emb_w[d] + emb_b[d]   (h lives in d_out)
// ---------------------------------------------------------------------------
__global__ void embed_kernel(const float* __restrict__ x,
                             const float* __restrict__ emb_w,
                             const float* __restrict__ emb_b,
                             float* __restrict__ h) {
    int i = blockIdx.x * blockDim.x + threadIdx.x;   // m*DMODEL + d
    if (i >= MROWS * DMODEL) return;
    int d = i & (DMODEL - 1);
    int m = i >> 8;
    h[i] = x[m] * emb_w[d] + emb_b[d];
}

// ---------------------------------------------------------------------------
// RMSNorm over D_MODEL=256, output bf16 (A operand of in_proj WMMA GEMM)
// ---------------------------------------------------------------------------
__global__ void rmsnorm_kernel(const float* __restrict__ h,
                               const float* __restrict__ w,
                               bf16_t* __restrict__ out) {
    __shared__ float red[256];
    int m = blockIdx.x;
    int d = threadIdx.x;
    float v = h[(size_t)m * DMODEL + d];
    red[d] = v * v;
    __syncthreads();
    for (int off = 128; off > 0; off >>= 1) {
        if (d < off) red[d] += red[d + off];
        __syncthreads();
    }
    float scale = rsqrtf(red[0] * (1.0f / DMODEL) + 1e-5f);
    out[(size_t)m * DMODEL + d] = (bf16_t)(v * scale * w[d]);
}

// ---------------------------------------------------------------------------
// WMMA bf16 GEMM:  Out[M,N] (+)= A[M,K] * W[N,K]^T,  f32 accumulate.
//
// Block = 256 threads (8 waves) -> 128 M-rows x (NT*16) N-cols of Out.
// Wave w owns M-tile rows [blockM*128 + w*16, +16) and all NT N-tiles,
// issuing NT back-to-back v_wmma_f32_16x16x32_bf16 per 32-K chunk with one
// shared A fragment (4x A reuse at NT=4).
//
// The weight chunk (NT*16 rows x 32 K, bf16) is shared by all 8 waves and is
// staged into LDS once per K-step with GLOBAL_LOAD_ASYNC_TO_LDS_B128
// (ASYNCcnt) when available, else a synchronous 16B copy.
//
// Fragment layouts per CDNA5 ISA 7.12.2:
//   A: lane L -> row M=L&15, K-chunks {8*(L>>4)+i, 16+8*(L>>4)+i}
//   B: lane L -> col N=L&15, K-chunks {8*(L>>4)+i, 16+8*(L>>4)+i}
//   D: vgpr r -> row m0 + 8*(L>>4) + r, col n0 + (L&15)
// EPI: 0 = store, 1 = accumulate (fused residual add)
// ---------------------------------------------------------------------------
template <int EPI, int NT>
__global__ void wmma_gemm_kernel(const bf16_t* __restrict__ A,
                                 const bf16_t* __restrict__ Wm,
                                 float* __restrict__ Out,
                                 int M, int N, int K, int NB) {
    __shared__ bf16_t sW[NT * 16 * 32];          // <= 4 KB

    const int blockM = blockIdx.x / NB;
    const int blockN = blockIdx.x - blockM * NB;
    const int wave   = threadIdx.x >> 5;
    const int lane   = threadIdx.x & 31;
    const int half   = lane >> 4;                 // 0 | 1
    const int r      = lane & 15;

    const int m0 = blockM * 128 + wave * 16;
    const bf16_t* arow = A  + (size_t)(m0 + r) * K;
    const bf16_t* wsup = Wm + (size_t)blockN * (NT * 16) * K;

    // staging assignment: thread t < NT*64 copies 16 bytes (8 bf16)
    const int srow = threadIdx.x >> 2;            // weight row in supertile
    const int skk  = (threadIdx.x & 3) * 8;       // K offset within chunk

    v8f acc[NT];
#pragma unroll
    for (int j = 0; j < NT; ++j) acc[j] = (v8f){};

    for (int k0 = 0; k0 < K; k0 += 32) {
        // ---- stage weight chunk into LDS ----
        if (threadIdx.x < NT * 64) {
            const bf16_t* src = wsup + (size_t)srow * K + k0 + skk;
            bf16_t* dst = &sW[srow * 32 + skk];
#ifdef HAVE_ASYNC_LDS
            v4i* srcg = (v4i*)const_cast<bf16_t*>(src);
            __builtin_amdgcn_global_load_async_to_lds_b128(
                (__attribute__((address_space(1))) v4i*)srcg,
                (__attribute__((address_space(3))) v4i*)dst, 0, 0);
#else
            *(v4i*)dst = *(const v4i*)src;
#endif
        }
#ifdef HAVE_ASYNC_LDS
#if __has_builtin(__builtin_amdgcn_s_wait_asynccnt)
        __builtin_amdgcn_s_wait_asynccnt(0);
#else
        asm volatile("s_wait_asynccnt 0" ::: "memory");
#endif
#endif
        __syncthreads();

        if (k0 + 32 < K)
            __builtin_prefetch(arow + k0 + 32, 0, 1);   // global_prefetch_b8

        // ---- A fragment (direct global, per wave) ----
        v16bf af;
#pragma unroll
        for (int i = 0; i < 8; ++i) {
            af[i]     = arow[k0 + 8 * half + i];        // K = kh .. kh+7
            af[8 + i] = arow[k0 + 16 + 8 * half + i];   // K = kh+16 .. kh+23
        }

        // ---- NT WMMAs sharing the A fragment ----
#pragma unroll
        for (int j = 0; j < NT; ++j) {
            const bf16_t* bp = &sW[(j * 16 + r) * 32 + 8 * half];
            v16bf bfr;
#pragma unroll
            for (int i = 0; i < 8; ++i) { bfr[i] = bp[i]; bfr[8 + i] = bp[16 + i]; }
            acc[j] = __builtin_amdgcn_wmma_f32_16x16x32_bf16(
                false, af, false, bfr, (short)0, acc[j], false, false);
        }
        __syncthreads();
    }

    // ---- epilogue ----
#pragma unroll
    for (int j = 0; j < NT; ++j) {
        int ncol = blockN * (NT * 16) + j * 16 + r;
#pragma unroll
        for (int i = 0; i < 8; ++i) {
            int mrow = m0 + 8 * half + i;
            size_t idx = (size_t)mrow * N + ncol;
            if (EPI == 1) Out[idx] += acc[j][i];
            else          Out[idx]  = acc[j][i];
        }
    }
}

// ---------------------------------------------------------------------------
// causal depthwise conv (k=4) over time + SiLU; bf16 (GEMM A) + f32 (scan u)
// ---------------------------------------------------------------------------
__global__ void conv_silu_kernel(const float* __restrict__ xz,   // [M][1024]
                                 const float* __restrict__ cw,   // [512][4]
                                 const float* __restrict__ cb,   // [512]
                                 bf16_t* __restrict__ xs_bf,
                                 float* __restrict__ xs_f) {
    int i = blockIdx.x * blockDim.x + threadIdx.x;   // m*512 + d
    if (i >= MROWS * DINNER) return;
    int d = i & (DINNER - 1);
    int m = i >> 9;
    int t = m & (SEQ - 1);
    float acc = cb[d];
#pragma unroll
    for (int j = 0; j < 4; ++j) {
        int tt = t - 3 + j;
        if (tt >= 0)
            acc += cw[d * 4 + j] * xz[(size_t)(m - 3 + j) * (2 * DINNER) + d];
    }
    float s = acc / (1.0f + __expf(-acc));           // silu
    xs_bf[i] = (bf16_t)s;
    xs_f[i]  = s;
}

// ---------------------------------------------------------------------------
// delta = softplus(dbc[:, :16] @ dtw^T + dtb) : K=16 -> VALU dot (BW-bound)
// ---------------------------------------------------------------------------
__global__ void delta_kernel(const float* __restrict__ dbc,  // [M][48]
                             const float* __restrict__ dtw,  // [512][16]
                             const float* __restrict__ dtb,  // [512]
                             float* __restrict__ delta) {
    int i = blockIdx.x * blockDim.x + threadIdx.x;   // m*512 + d
    if (i >= MROWS * DINNER) return;
    int d = i & (DINNER - 1);
    int m = i >> 9;
    float acc = dtb[d];
#pragma unroll
    for (int r = 0; r < DTRANK; ++r)
        acc += dbc[(size_t)m * DBCDIM + r] * dtw[d * DTRANK + r];
    delta[i] = (acc > 20.0f) ? acc : log1pf(__expf(acc));
}

// ---------------------------------------------------------------------------
// selective scan (sequential over SEQ) + SiLU(z) gate, bf16 out for out_proj.
// 8 blocks x 256 threads; one thread = one (b, channel); 16 states in regs.
// ---------------------------------------------------------------------------
__global__ void scan_kernel(const float* __restrict__ delta, // [M][512]
                            const float* __restrict__ xs_f,  // [M][512]
                            const float* __restrict__ dbc,   // [M][48]
                            const float* __restrict__ xz,    // [M][1024]
                            const float* __restrict__ A_log, // [512][16]
                            const float* __restrict__ Dp,    // [512]
                            bf16_t* __restrict__ ygate) {    // [M][512]
    __shared__ float sB[DSTATE];
    __shared__ float sC[DSTATE];
    int g  = blockIdx.x * blockDim.x + threadIdx.x;
    int b  = g >> 9;
    int ch = g & (DINNER - 1);

    float Av[DSTATE];
#pragma unroll
    for (int s = 0; s < DSTATE; ++s)
        Av[s] = -__expf(A_log[ch * DSTATE + s]);
    float Dv = Dp[ch];

    float hst[DSTATE];
#pragma unroll
    for (int s = 0; s < DSTATE; ++s) hst[s] = 0.0f;

    for (int t = 0; t < SEQ; ++t) {
        int m = b * SEQ + t;
        if (threadIdx.x < 2 * DSTATE) {
            float v = dbc[(size_t)m * DBCDIM + DTRANK + threadIdx.x];
            if (threadIdx.x < DSTATE) sB[threadIdx.x] = v;
            else                      sC[threadIdx.x - DSTATE] = v;
        }
        __syncthreads();

        float dl = delta[(size_t)m * DINNER + ch];
        float u  = xs_f[(size_t)m * DINNER + ch];
        float y  = u * Dv;
#pragma unroll
        for (int s = 0; s < DSTATE; ++s) {
            float dA = __expf(dl * Av[s]);
            hst[s] = dA * hst[s] + (dl * sB[s]) * u;
            y += hst[s] * sC[s];
        }
        float z = xz[(size_t)m * (2 * DINNER) + DINNER + ch];
        y *= z / (1.0f + __expf(-z));                 // * silu(z)
        ygate[(size_t)m * DINNER + ch] = (bf16_t)y;
        __syncthreads();
    }
}

// ---------------------------------------------------------------------------
// host launch
// ---------------------------------------------------------------------------
extern "C" void kernel_launch(void* const* d_in, const int* in_sizes, int n_in,
                              void* d_out, int out_size, void* d_ws, size_t ws_size,
                              hipStream_t stream) {
    const float* x       = (const float*)d_in[0];
    const float* emb_w   = (const float*)d_in[1];
    const float* emb_b   = (const float*)d_in[2];
    const float* in_w    = (const float*)d_in[3];   // [2][1024][256]
    const float* conv_w  = (const float*)d_in[4];   // [2][512][4]
    const float* conv_b  = (const float*)d_in[5];   // [2][512]
    const float* xp_w    = (const float*)d_in[6];   // [2][48][512]
    const float* dt_w    = (const float*)d_in[7];   // [2][512][16]
    const float* dt_b    = (const float*)d_in[8];   // [2][512]
    const float* A_log   = (const float*)d_in[9];   // [2][512][16]
    const float* Dp      = (const float*)d_in[10];  // [2][512]
    const float* out_w   = (const float*)d_in[11];  // [2][256][512]
    const float* norm_w  = (const float*)d_in[12];  // [2][256]

    float* hbuf = (float*)d_out;                    // h lives here (8192 x 256)

    char* ws = (char*)d_ws;
    size_t off = 0;
    auto carve = [&](size_t bytes) -> void* {
        void* p = ws + off;
        off = (off + bytes + 255) & ~(size_t)255;
        return p;
    };

    float*  xz      = (float*)carve((size_t)MROWS * 2 * DINNER * 4);  // 32 MB
    bf16_t* normed  = (bf16_t*)carve((size_t)MROWS * DMODEL * 2);     //  4 MB
    bf16_t* xs_bf   = (bf16_t*)carve((size_t)MROWS * DINNER * 2);     //  8 MB
    float*  xs_f    = (float*)carve((size_t)MROWS * DINNER * 4);      // 16 MB
    float*  dbc     = (float*)carve((size_t)MROWS * DBCDIM * 4);      // 1.5 MB
    float*  delta   = (float*)carve((size_t)MROWS * DINNER * 4);      // 16 MB
    bf16_t* ygate   = (bf16_t*)carve((size_t)MROWS * DINNER * 2);     //  8 MB
    bf16_t* in_wbf  = (bf16_t*)carve((size_t)2 * 2 * DINNER * DMODEL * 2);
    bf16_t* xp_wbf  = (bf16_t*)carve((size_t)2 * DBCDIM * DINNER * 2);
    bf16_t* out_wbf = (bf16_t*)carve((size_t)2 * DMODEL * DINNER * 2);

    // weight conversion (both layers at once; weights contiguous)
    {
        int n1 = 2 * 2 * DINNER * DMODEL;   // in_proj
        int n2 = 2 * DBCDIM * DINNER;       // x_proj
        int n3 = 2 * DMODEL * DINNER;       // out_proj
        cvt_bf16_kernel<<<(n1 + 255) / 256, 256, 0, stream>>>(in_w,  in_wbf,  n1);
        cvt_bf16_kernel<<<(n2 + 255) / 256, 256, 0, stream>>>(xp_w,  xp_wbf,  n2);
        cvt_bf16_kernel<<<(n3 + 255) / 256, 256, 0, stream>>>(out_w, out_wbf, n3);
    }

    // embedding
    embed_kernel<<<(MROWS * DMODEL + 255) / 256, 256, 0, stream>>>(x, emb_w, emb_b, hbuf);

    for (int l = 0; l < 2; ++l) {
        // 1) RMSNorm -> bf16
        rmsnorm_kernel<<<MROWS, 256, 0, stream>>>(hbuf, norm_w + l * DMODEL, normed);

        // 2) in_proj: xz = normed @ in_w^T   (8192 x 1024, K=256), NT=4
        {
            int NB = (2 * DINNER) / 64;                 // 16
            wmma_gemm_kernel<0, 4><<<(MROWS / 128) * NB, 256, 0, stream>>>(
                normed, in_wbf + (size_t)l * 2 * DINNER * DMODEL, xz,
                MROWS, 2 * DINNER, DMODEL, NB);
        }

        // 3) depthwise causal conv + silu
        conv_silu_kernel<<<(MROWS * DINNER + 255) / 256, 256, 0, stream>>>(
            xz, conv_w + (size_t)l * DINNER * 4, conv_b + (size_t)l * DINNER,
            xs_bf, xs_f);

        // 4) x_proj: dbc = xs @ xp_w^T   (8192 x 48, K=512), NT=3
        {
            int NB = DBCDIM / 48;                       // 1
            wmma_gemm_kernel<0, 3><<<(MROWS / 128) * NB, 256, 0, stream>>>(
                xs_bf, xp_wbf + (size_t)l * DBCDIM * DINNER, dbc,
                MROWS, DBCDIM, DINNER, NB);
        }

        // 5) delta = softplus(dbc[:, :16] @ dt_w^T + dt_b)
        delta_kernel<<<(MROWS * DINNER + 255) / 256, 256, 0, stream>>>(
            dbc, dt_w + (size_t)l * DINNER * DTRANK, dt_b + (size_t)l * DINNER, delta);

        // 6) selective scan + silu(z) gate
        scan_kernel<<<(BATCH * DINNER) / 256, 256, 0, stream>>>(
            delta, xs_f, dbc, xz,
            A_log + (size_t)l * DINNER * DSTATE, Dp + (size_t)l * DINNER, ygate);

        // 7) out_proj with fused residual: h += ygate @ out_w^T (8192x256,K=512), NT=4
        {
            int NB = DMODEL / 64;                       // 4
            wmma_gemm_kernel<1, 4><<<(MROWS / 128) * NB, 256, 0, stream>>>(
                ygate, out_wbf + (size_t)l * DMODEL * DINNER, hbuf,
                MROWS, DMODEL, DINNER, NB);
        }
    }
}
